// MultiHeadSelfAttention_541165879309
// MI455X (gfx1250) — matrix operand order repaired and checked by reading, not compiled, not run on hardware
//
#include <hip/hip_runtime.h>

// ---------------------------------------------------------------------------
// MHSA forward for MI455X (gfx1250, wave32, WMMA + async global->LDS DMA).
// B=2, S=2048, E=1024, H=16, HD=64.
// Pipeline: convert x/W to f16 (W transposed) once; GEMMs run
// v_wmma_f32_16x16x32_f16 with DOUBLE-BUFFERED 64-deep k-tiles staged by
// global_load_async_to_lds_b128 (ASYNCcnt, counted waits), last tile peeled.
// ---------------------------------------------------------------------------

typedef _Float16 half_t;
typedef _Float16 v16h __attribute__((ext_vector_type(16)));
typedef _Float16 v8h  __attribute__((ext_vector_type(8)));
typedef _Float16 v4h  __attribute__((ext_vector_type(4)));
typedef float    v8f  __attribute__((ext_vector_type(8)));
typedef float    v4f  __attribute__((ext_vector_type(4)));

#define B_    2
#define S_    2048
#define E_    1024
#define H_    16
#define HD_   64
#define N3E_  3072
#define MROWS (B_ * S_)     // 4096
#define SCALE_ 0.125f       // 1/sqrt(64)
#define LDP   40            // attention LDS pitch in halves (80B)
#define LDPW  72            // GEMM LDS pitch in halves (144B, 16B-aligned)

static __device__ __forceinline__ v16h cat16(v8h lo, v8h hi) {
  return __builtin_shufflevector(lo, hi, 0,1,2,3,4,5,6,7,8,9,10,11,12,13,14,15);
}
static __device__ __forceinline__ v8f wmma16(v16h a, v16h b, v8f c) {
  return __builtin_amdgcn_wmma_f32_16x16x32_f16(false, a, false, b, (short)0, c,
                                                false, false);
}

// LDS byte-offset of a __shared__ object (generic -> addrspace(3) -> u32).
static __device__ __forceinline__ unsigned lds_off(void* p) {
  return (unsigned)(unsigned long long)(__attribute__((address_space(3))) char*)p;
}
// Async DMA: 16 bytes global -> LDS per lane (ASYNCcnt-tracked, GV mode).
static __device__ __forceinline__ void async_b128(unsigned lds, const void* g) {
  asm volatile("global_load_async_to_lds_b128 %0, %1, off"
               :: "v"(lds), "v"((unsigned long long)(size_t)g)
               : "memory");
}
static __device__ __forceinline__ void wait_async0() {
  asm volatile("s_wait_asynccnt 0x0" ::: "memory");
}
static __device__ __forceinline__ void wait_async6() {
  asm volatile("s_wait_asynccnt 0x6" ::: "memory");
}

// ---------------------------------------------------------------------------
// Pre-pass: elementwise f32 -> f16 (for x).
// ---------------------------------------------------------------------------
__global__ __launch_bounds__(256)
void cvt_f32_f16_kernel(const float* __restrict__ in, half_t* __restrict__ out) {
  const size_t i = ((size_t)blockIdx.x * 256 + threadIdx.x) * 4;
  v4f f = *(const v4f*)(in + i);
  v4h h;
#pragma unroll
  for (int j = 0; j < 4; ++j) h[j] = (half_t)f[j];
  *(v4h*)(out + i) = h;
}

// ---------------------------------------------------------------------------
// Pre-pass: W [K][N] f32 -> W^T [N][K] f16 (LDS-tiled 32x32 transpose).
// ---------------------------------------------------------------------------
__global__ __launch_bounds__(256)
void transpose_cvt_kernel(const float* __restrict__ in, half_t* __restrict__ outT,
                          int K, int N) {
  __shared__ float tile[32][33];
  const int k0 = blockIdx.y * 32, n0 = blockIdx.x * 32;
  const int tx = threadIdx.x & 31, ty = threadIdx.x >> 5;
#pragma unroll
  for (int i = 0; i < 4; ++i)
    tile[ty + 8 * i][tx] = in[(size_t)(k0 + ty + 8 * i) * N + n0 + tx];
  __syncthreads();
#pragma unroll
  for (int i = 0; i < 4; ++i)
    outT[(size_t)(n0 + ty + 8 * i) * K + k0 + tx] = (half_t)tile[tx][ty + 8 * i];
}

// ---------------------------------------------------------------------------
// Shared GEMM machinery: block 256 thr (8 waves), tile M=64 x N=128,
// wave tile 32x32, k-step 64 (8 WMMAs/step), double-buffered async staging.
// A [M][K] f16 row-major, BT [N][K] f16 row-major (pre-transposed weights).
// ---------------------------------------------------------------------------
struct GemmCtx {
  int tid, lane, w, wm, wn, hi, l16, ar, ac, br, bc;
};
static __device__ __forceinline__ GemmCtx gemm_ctx() {
  GemmCtx c;
  c.tid = threadIdx.x;
  c.lane = c.tid & 31; c.w = c.tid >> 5;
  c.wm = c.w & 1;      c.wn = c.w >> 1;          // 2x4 wave grid
  c.hi = c.lane >> 4;  c.l16 = c.lane & 15;
  c.ar = c.tid >> 2;   c.ac = (c.tid & 3) * 16;  // A: 64 rows x 4x16 halves
  c.br = c.tid >> 1;   c.bc = (c.tid & 1) * 32;  // B: 128 rows x 2x32 halves
  return c;
}
// 6 async b128s per thread per 64-deep tile.
static __device__ __forceinline__ void gemm_issue(
    const GemmCtx& c, const half_t* __restrict__ A, const half_t* __restrict__ BT,
    half_t (*As)[LDPW], half_t (*Bs)[LDPW], int row0, int col0, int k0) {
  const half_t* ag = A  + (size_t)(row0 + c.ar) * E_ + k0 + c.ac;
  const half_t* bg = BT + (size_t)(col0 + c.br) * E_ + k0 + c.bc;
  async_b128(lds_off(&As[c.ar][c.ac]),      ag);
  async_b128(lds_off(&As[c.ar][c.ac + 8]),  ag + 8);
  async_b128(lds_off(&Bs[c.br][c.bc]),      bg);
  async_b128(lds_off(&Bs[c.br][c.bc + 8]),  bg + 8);
  async_b128(lds_off(&Bs[c.br][c.bc + 16]), bg + 16);
  async_b128(lds_off(&Bs[c.br][c.bc + 24]), bg + 24);
}
static __device__ __forceinline__ void gemm_step(
    const GemmCtx& c, half_t (*As)[LDPW], half_t (*Bs)[LDPW], v8f acc[2][2]) {
#pragma unroll
  for (int kk = 0; kk < 64; kk += 32) {
    v16h af[2];
#pragma unroll
    for (int mt = 0; mt < 2; ++mt) {
      const int m = c.wm * 32 + mt * 16 + c.l16;
      af[mt] = cat16(*(const v8h*)&As[m][kk + c.hi * 8],
                     *(const v8h*)&As[m][kk + 16 + c.hi * 8]);
    }
#pragma unroll
    for (int nt = 0; nt < 2; ++nt) {
      const int n = c.wn * 32 + nt * 16 + c.l16;
      v16h bf = cat16(*(const v8h*)&Bs[n][kk + c.hi * 8],
                      *(const v8h*)&Bs[n][kk + 16 + c.hi * 8]);
#pragma unroll
      for (int mt = 0; mt < 2; ++mt)
        acc[mt][nt] = wmma16(af[mt], bf, acc[mt][nt]);
    }
  }
}

// ---------------------------------------------------------------------------
// Kernel 1: QKV GEMM; epilogue scatters Q,K as [B,H,S,HD], V as [B,H,HD,S].
// ---------------------------------------------------------------------------
__global__ __launch_bounds__(256)
void qkv_gemm_kernel(const half_t* __restrict__ A, const half_t* __restrict__ BT,
                     const float* __restrict__ bias,
                     half_t* __restrict__ Qo, half_t* __restrict__ Ko,
                     half_t* __restrict__ Vo) {
  __shared__ __align__(16) half_t As[2][64][LDPW];
  __shared__ __align__(16) half_t Bs[2][128][LDPW];

  const GemmCtx c = gemm_ctx();
  const int row0 = blockIdx.y * 64;
  const int col0 = blockIdx.x * 128;

  const v8f vzero = {0.f,0.f,0.f,0.f,0.f,0.f,0.f,0.f};
  v8f acc[2][2] = {{vzero, vzero}, {vzero, vzero}};

  gemm_issue(c, A, BT, As[0], Bs[0], row0, col0, 0);
  int k0 = 0;
  for (; k0 < E_ - 64; k0 += 64) {         // steady state: branch-free body
    const int cur = (k0 >> 6) & 1;
    gemm_issue(c, A, BT, As[cur ^ 1], Bs[cur ^ 1], row0, col0, k0 + 64);
    wait_async6();                         // current tile landed (in-order)
    __syncthreads();
    gemm_step(c, As[cur], Bs[cur], acc);
    __syncthreads();                       // reads done before next overwrite
  }
  {                                        // peeled last tile
    const int cur = (k0 >> 6) & 1;
    wait_async0();
    __syncthreads();
    gemm_step(c, As[cur], Bs[cur], acc);
  }

#pragma unroll
  for (int nt = 0; nt < 2; ++nt) {
    const int n   = col0 + c.wn * 32 + nt * 16 + c.l16;
    const int h   = n / 192;
    const int rem = n % 192;
    const int whc = rem >> 6;              // 0=Q 1=K 2=V
    const int d   = rem & 63;
    const float bv = bias[n];
#pragma unroll
    for (int mt = 0; mt < 2; ++mt) {
#pragma unroll
      for (int r = 0; r < 8; ++r) {
        const int m = row0 + c.wm * 32 + mt * 16 + r + 8 * c.hi;
        const int b = m >> 11, s = m & 2047;
        const half_t val = (half_t)(acc[mt][nt][r] + bv);
        const size_t bh = (size_t)(b * H_ + h);
        if (whc == 0)      Qo[(bh * S_ + s) * HD_ + d] = val;
        else if (whc == 1) Ko[(bh * S_ + s) * HD_ + d] = val;
        else               Vo[(bh * HD_ + d) * S_ + s] = val;
      }
    }
  }
}

// ---------------------------------------------------------------------------
// Kernel 2: flash attention.  4 waves/block, one 16-row Q tile per wave.
// V tile (64 d x 32 kv) async-DMA'd to LDS, overlapped with QK^T + softmax;
// next K tile prefetched while scoring the current one.
// ---------------------------------------------------------------------------
__global__ __launch_bounds__(128)
void attn_kernel(const half_t* __restrict__ Q, const half_t* __restrict__ K,
                 const half_t* __restrict__ Vt, half_t* __restrict__ Aout) {
  __shared__ __align__(16) half_t Vs[64][LDP];       // [d][kv]
  __shared__ __align__(16) half_t P[4][16][LDP];     // per-wave probs [m][kv]

  const int tid  = threadIdx.x;
  const int lane = tid & 31, w = tid >> 5;
  const int hi = lane >> 4, l16 = lane & 15;
  const int bh = blockIdx.y;
  const int b = bh >> 4, h = bh & 15;
  const int q0 = blockIdx.x * 64 + w * 16;

  const half_t* Qb = Q  + (size_t)bh * S_ * HD_;
  const half_t* Kb = K  + (size_t)bh * S_ * HD_;
  const half_t* Vb = Vt + (size_t)bh * HD_ * S_;     // [d][s]

  const half_t* qrow = Qb + (size_t)(q0 + l16) * HD_;
  const v16h aq0 = cat16(*(const v8h*)(qrow + hi * 8),
                         *(const v8h*)(qrow + 16 + hi * 8));
  const v16h aq1 = cat16(*(const v8h*)(qrow + 32 + hi * 8),
                         *(const v8h*)(qrow + 48 + hi * 8));

  const v8f vzero = {0.f,0.f,0.f,0.f,0.f,0.f,0.f,0.f};
  v8f acc[4] = {vzero, vzero, vzero, vzero};
  float mrow[8], lrow[8];
#pragma unroll
  for (int r = 0; r < 8; ++r) { mrow[r] = -1e30f; lrow[r] = 0.f; }

  for (int kv0 = 0; kv0 < S_; kv0 += 32) {
    __syncthreads();                 // protect Vs reuse across iterations
    // async DMA V tile: rows d, 32 kv halves each (2 x b128 per thread)
#pragma unroll
    for (int j = 0; j < 2; ++j) {
      const int cch = tid * 2 + j;   // chunk 0..255
      const int d = cch >> 2, seg = (cch & 3) * 8;
      async_b128(lds_off(&Vs[d][seg]), Vb + (size_t)d * S_ + kv0 + seg);
    }
    // prefetch next K tile rows while scoring this one
    if (kv0 + 32 < S_) {
      __builtin_prefetch(Kb + (size_t)(kv0 + 32 + l16) * HD_, 0, 0);
      __builtin_prefetch(Kb + (size_t)(kv0 + 48 + l16) * HD_, 0, 0);
    }

    // scores (overlaps the async V copy)
    v8f c0 = vzero, c1 = vzero;
    {
      const half_t* kr0 = Kb + (size_t)(kv0 + l16) * HD_;
      v16h bk = cat16(*(const v8h*)(kr0 + hi * 8),
                      *(const v8h*)(kr0 + 16 + hi * 8));
      c0 = wmma16(aq0, bk, c0);
      bk = cat16(*(const v8h*)(kr0 + 32 + hi * 8),
                 *(const v8h*)(kr0 + 48 + hi * 8));
      c0 = wmma16(aq1, bk, c0);
      const half_t* kr1 = Kb + (size_t)(kv0 + 16 + l16) * HD_;
      bk = cat16(*(const v8h*)(kr1 + hi * 8),
                 *(const v8h*)(kr1 + 16 + hi * 8));
      c1 = wmma16(aq0, bk, c1);
      bk = cat16(*(const v8h*)(kr1 + 32 + hi * 8),
                 *(const v8h*)(kr1 + 48 + hi * 8));
      c1 = wmma16(aq1, bk, c1);
    }

    // online softmax; C layout: lane holds key col l16(+16), rows r + 8*hi
    float alpha[8];
#pragma unroll
    for (int r = 0; r < 8; ++r) {
      const float s0 = c0[r] * SCALE_;
      const float s1 = c1[r] * SCALE_;
      float tm = fmaxf(s0, s1);
      tm = fmaxf(tm, __shfl_xor(tm, 1));
      tm = fmaxf(tm, __shfl_xor(tm, 2));
      tm = fmaxf(tm, __shfl_xor(tm, 4));
      tm = fmaxf(tm, __shfl_xor(tm, 8));
      const float mn = fmaxf(mrow[r], tm);
      const float al = __expf(mrow[r] - mn);
      mrow[r] = mn;
      const float p0 = __expf(s0 - mn);
      const float p1 = __expf(s1 - mn);
      float ts = p0 + p1;
      ts += __shfl_xor(ts, 1);
      ts += __shfl_xor(ts, 2);
      ts += __shfl_xor(ts, 4);
      ts += __shfl_xor(ts, 8);
      lrow[r] = lrow[r] * al + ts;
      alpha[r] = al;
      const int m = r + 8 * hi;
      P[w][m][l16]      = (half_t)p0;
      P[w][m][16 + l16] = (half_t)p1;
    }
#pragma unroll
    for (int t = 0; t < 4; ++t)
#pragma unroll
      for (int r = 0; r < 8; ++r) acc[t][r] *= alpha[r];

    wait_async0();                   // V tile landed
    __syncthreads();                 // all waves: V + own P visible

    const v16h apf = cat16(*(const v8h*)&P[w][l16][hi * 8],
                           *(const v8h*)&P[w][l16][16 + hi * 8]);
#pragma unroll
    for (int t = 0; t < 4; ++t) {
      const int n = t * 16 + l16;
      v16h bv = cat16(*(const v8h*)&Vs[n][hi * 8],
                      *(const v8h*)&Vs[n][16 + hi * 8]);
      acc[t] = wmma16(apf, bv, acc[t]);
    }
  }

  float inv[8];
#pragma unroll
  for (int r = 0; r < 8; ++r) inv[r] = 1.f / lrow[r];
#pragma unroll
  for (int t = 0; t < 4; ++t) {
#pragma unroll
    for (int r = 0; r < 8; ++r) {
      const int s = q0 + r + 8 * hi;
      const size_t g = (size_t)(b * S_ + s);
      Aout[g * E_ + h * HD_ + t * 16 + l16] = (half_t)(acc[t][r] * inv[r]);
    }
  }
}

// ---------------------------------------------------------------------------
// Kernel 3: output projection.  A = attn f16 [M][K], BT = Wout^T f16 [N][K].
// ---------------------------------------------------------------------------
__global__ __launch_bounds__(256)
void out_gemm_kernel(const half_t* __restrict__ A, const half_t* __restrict__ BT,
                     const float* __restrict__ bias, float* __restrict__ out) {
  __shared__ __align__(16) half_t As[2][64][LDPW];
  __shared__ __align__(16) half_t Bs[2][128][LDPW];

  const GemmCtx c = gemm_ctx();
  const int row0 = blockIdx.y * 64;
  const int col0 = blockIdx.x * 128;

  const v8f vzero = {0.f,0.f,0.f,0.f,0.f,0.f,0.f,0.f};
  v8f acc[2][2] = {{vzero, vzero}, {vzero, vzero}};

  gemm_issue(c, A, BT, As[0], Bs[0], row0, col0, 0);
  int k0 = 0;
  for (; k0 < E_ - 64; k0 += 64) {
    const int cur = (k0 >> 6) & 1;
    gemm_issue(c, A, BT, As[cur ^ 1], Bs[cur ^ 1], row0, col0, k0 + 64);
    wait_async6();
    __syncthreads();
    gemm_step(c, As[cur], Bs[cur], acc);
    __syncthreads();
  }
  {
    const int cur = (k0 >> 6) & 1;
    wait_async0();
    __syncthreads();
    gemm_step(c, As[cur], Bs[cur], acc);
  }

#pragma unroll
  for (int nt = 0; nt < 2; ++nt) {
    const int n = col0 + c.wn * 32 + nt * 16 + c.l16;
    const float bv = bias[n];
#pragma unroll
    for (int mt = 0; mt < 2; ++mt) {
#pragma unroll
      for (int r = 0; r < 8; ++r) {
        const int m = row0 + c.wm * 32 + mt * 16 + r + 8 * c.hi;
        out[(size_t)m * E_ + n] = acc[mt][nt][r] + bv;
      }
    }
  }
}

// ---------------------------------------------------------------------------
extern "C" void kernel_launch(void* const* d_in, const int* in_sizes, int n_in,
                              void* d_out, int out_size, void* d_ws,
                              size_t ws_size, hipStream_t stream) {
  (void)in_sizes; (void)n_in; (void)out_size; (void)ws_size;
  const float* x    = (const float*)d_in[0];
  const float* Wqkv = (const float*)d_in[1];
  const float* bqkv = (const float*)d_in[2];
  const float* Wout = (const float*)d_in[3];
  const float* bout = (const float*)d_in[4];
  float* out = (float*)d_out;

  char* ws = (char*)d_ws;
  const size_t MB = 1024 * 1024;
  half_t* x16    = (half_t*)(ws);              // 8 MB  [M][E]
  half_t* WqkvT  = (half_t*)(ws + 8 * MB);     // 6 MB  [3E][E]
  half_t* WoutT  = (half_t*)(ws + 14 * MB);    // 2 MB  [E][E]
  half_t* Qf     = (half_t*)(ws + 16 * MB);    // 8 MB  [B,H,S,HD]
  half_t* Kf     = (half_t*)(ws + 24 * MB);    // 8 MB  [B,H,S,HD]
  half_t* Vtf    = (half_t*)(ws + 32 * MB);    // 8 MB  [B,H,HD,S]
  half_t* Af     = (half_t*)(ws + 40 * MB);    // 8 MB  [B,S,E]

  cvt_f32_f16_kernel<<<(MROWS * E_) / (256 * 4), 256, 0, stream>>>(x, x16);
  transpose_cvt_kernel<<<dim3(N3E_ / 32, E_ / 32), 256, 0, stream>>>(
      Wqkv, WqkvT, E_, N3E_);
  transpose_cvt_kernel<<<dim3(E_ / 32, E_ / 32), 256, 0, stream>>>(
      Wout, WoutT, E_, E_);

  qkv_gemm_kernel<<<dim3(N3E_ / 128, MROWS / 64), 256, 0, stream>>>(
      x16, WqkvT, bqkv, Qf, Kf, Vtf);
  attn_kernel<<<dim3(S_ / 64, B_ * H_), 128, 0, stream>>>(Qf, Kf, Vtf, Af);
  out_gemm_kernel<<<dim3(E_ / 128, MROWS / 64), 256, 0, stream>>>(
      Af, WoutT, bout, out);
}
